// Generator_60559038874150
// MI455X (gfx1250) — compile-verified, matrix operand order checked by reading, not run on hardware
//
// Contextual attention (DeepFill) for MI455X / gfx1250 — compile-only iteration.
//
// Pipeline (all per-call, deterministic, stream-ordered):
//   zero(y) -> eq(mask) -> X_f16 patches(fg) -> Wn_f16 normalized patches(bg)
//   -> score GEMM (v_wmma_f32_16x16x32_f16, M=N=1024,K=1152 per batch)
//   -> fuse1 (diag 3-tap) -> fuse2 (transposed diag 3-tap, index-exact vs ref)
//   -> masked softmax over patch axis -> P^T f16
//   -> deconv GEMM (WMMA, M=128,N=1024,K=1024 per (b,dy,dx)) with
//      global_atomic_add_f32 overlap-add epilogue, /4 scale.
//
// Double-buffered LDS staging driven by GLOBAL_LOAD_ASYNC_TO_LDS_B128
// (ASYNCcnt + s_wait_asynccnt + barrier), overlapping next-tile fill with the
// current tile's 8 WMMAs. Builtin takes int4 addrspace(1)* / addrspace(3)*
// (per hipcc diagnostic); guarded by __has_builtin with sync fallback.
//
// Precision: f16 operands, f32 accumulation (reference is f32; f16 WMMA keeps
// the 54-GFLOP problem memory-bound at 23.3 TB/s instead of f32-matrix bound).
// Workspace requirement: ~105 MB (offsets below, buffers reused across stages).

#include <hip/hip_runtime.h>
#include <hip/hip_bf16.h>
#include <math.h>
#include <stdint.h>

typedef _Float16 half_t;
typedef __attribute__((ext_vector_type(8)))  _Float16 v8h;
typedef __attribute__((ext_vector_type(16))) _Float16 v16h;
typedef __attribute__((ext_vector_type(8)))  float    v8f;
typedef int v4i __attribute__((__vector_size__(16)));
typedef __attribute__((address_space(1))) v4i* as1_v4i;
typedef __attribute__((address_space(3))) v4i* as3_v4i;

#define KDIM 1152   // C*9
#define LPAT 1024   // 32*32 patch positions
#define LDSW 40     // 32 K halves + 8 pad (rows stay 16B aligned: 80B stride)

#if __has_builtin(__builtin_amdgcn_global_load_async_to_lds_b128) && \
    __has_builtin(__builtin_amdgcn_s_wait_asynccnt)
#define USE_ASYNC_LDS 1
#else
#define USE_ASYNC_LDS 0
#endif

// 16B global->LDS copy. Async path: GLOBAL_LOAD_ASYNC_TO_LDS_B128 (ASYNCcnt,
// no VGPR round trip). LDS generic addr low 32 bits == LDS offset (aperture
// rule), so the integer-detour cast yields a valid addrspace(3) pointer.
__device__ __forceinline__ void cp16(half_t* l, const half_t* g) {
#if USE_ASYNC_LDS
  __builtin_amdgcn_global_load_async_to_lds_b128(
      (as1_v4i)(uintptr_t)g,
      (as3_v4i)(uint32_t)(uintptr_t)l, 0, 0);
#else
  *(v8h*)l = *(const v8h*)g;
#endif
}
__device__ __forceinline__ void async_wait0() {
#if USE_ASYNC_LDS
  __builtin_amdgcn_s_wait_asynccnt(0);
#endif
}

union FragU { v16h v; v8h h[2]; };

// CDNA5 16-bit A/B fragment layout (ISA 7.12.2): lane L holds row (L&15);
// halves 0..7 -> K = base + (L>=16?8:0) + 0..7 ; halves 8..15 -> +16.
__device__ __forceinline__ v16h ldsFrag(const half_t* tile, int row0, int lane) {
  const half_t* rp = tile + (row0 + (lane & 15)) * LDSW;
  int ko = (lane & 16) ? 8 : 0;
  FragU u;
  u.h[0] = *(const v8h*)(rp + ko);
  u.h[1] = *(const v8h*)(rp + 16 + ko);
  return u.v;
}

// ---------------------------------------------------------------- utilities
__global__ void zero_f32(float* __restrict__ p, int n) {
  int i = blockIdx.x * 256 + threadIdx.x;
  if (i < n) p[i] = 0.f;
}

// eq[p] = 1 if the 3x3 (zero-padded) window of the downsampled mask sums to 0
__global__ void eq_kernel(const float* __restrict__ mask, float* __restrict__ eq) {
  int p = blockIdx.x * 256 + threadIdx.x;
  if (p >= LPAT) return;
  int py = p >> 5, px = p & 31;
  float s = 0.f;
  for (int dy = 0; dy < 3; ++dy)
    for (int dx = 0; dx < 3; ++dx) {
      int yy = py + dy - 1, xx = px + dx - 1;
      if (yy >= 0 && yy < 32 && xx >= 0 && xx < 32)
        s += mask[(2 * yy) * 64 + 2 * xx];
    }
  eq[p] = (s == 0.f) ? 1.f : 0.f;
}

// X[b][f][c*9 + dy*3 + dx] = fg_ds padded 3x3 patch, f16
__global__ void __launch_bounds__(128) x_prep(const float* __restrict__ fg,
                                              half_t* __restrict__ X) {
  int f = blockIdx.x, b = blockIdx.y, c = threadIdx.x;
  int yf = f >> 5, xf = f & 31;
  size_t base = (size_t)(b * LPAT + f) * KDIM + (size_t)c * 9;
  for (int r = 0; r < 9; ++r) {
    int dy = r / 3, dx = r - 3 * (r / 3);
    int yy = yf + dy - 1, xx = xf + dx - 1;
    float v = 0.f;
    if (yy >= 0 && yy < 32 && xx >= 0 && xx < 32)
      v = fg[((size_t)(b * 128 + c) * 64 + 2 * yy) * 64 + 2 * xx];
    X[base + r] = (half_t)v;
  }
}

// Wn[b][p][k] = bg_ds patch / max(||patch||_2, 1e-4), f16
__global__ void __launch_bounds__(128) wn_prep(const float* __restrict__ bg,
                                               half_t* __restrict__ Wn) {
  __shared__ float red[128];
  int p = blockIdx.x, b = blockIdx.y, c = threadIdx.x;
  int py = p >> 5, px = p & 31;
  float w[9], ss = 0.f;
  for (int r = 0; r < 9; ++r) {
    int dy = r / 3, dx = r - 3 * (r / 3);
    int yy = py + dy - 1, xx = px + dx - 1;
    float v = 0.f;
    if (yy >= 0 && yy < 32 && xx >= 0 && xx < 32)
      v = bg[((size_t)(b * 128 + c) * 64 + 2 * yy) * 64 + 2 * xx];
    w[r] = v; ss += v * v;
  }
  red[c] = ss; __syncthreads();
  for (int s = 64; s > 0; s >>= 1) {
    if (c < s) red[c] += red[c + s];
    __syncthreads();
  }
  float inv = 1.f / fmaxf(sqrtf(red[0]), 1e-4f);
  size_t base = (size_t)(b * LPAT + p) * KDIM + (size_t)c * 9;
  for (int r = 0; r < 9; ++r) Wn[base + r] = (half_t)(w[r] * inv);
}

// ------------------------------------------------ score GEMM: S = Wn * X^T
// block tile 128x128x32, 8 waves (4M x 2N), 32x64 per wave -> 2x4 WMMA accums
// double-buffered LDS; tile i+1 streams in (async) while tile i does 8 WMMAs
__global__ void __launch_bounds__(256) score_gemm(const half_t* __restrict__ Wn,
                                                  const half_t* __restrict__ X,
                                                  float* __restrict__ S) {
  __shared__ __align__(16) half_t As[2][128 * LDSW];
  __shared__ __align__(16) half_t Bs[2][128 * LDSW];
  int tid = threadIdx.x, lane = tid & 31, wave = tid >> 5;
  int wm = wave & 3, wn_ = wave >> 2;
  int b = blockIdx.z, pBase = blockIdx.y * 128, fBase = blockIdx.x * 128;
  const half_t* Ag = Wn + (size_t)(b * LPAT + pBase) * KDIM;
  const half_t* Bg = X  + (size_t)(b * LPAT + fBase) * KDIM;
  int rs = tid >> 2, sg = (tid & 3) << 3;       // this thread's 16B copy units
  v8f acc[2][4];
  for (int mi = 0; mi < 2; ++mi)
    for (int nj = 0; nj < 4; ++nj) acc[mi][nj] = {};

  // prologue: stage tile 0 into buffer 0
  for (int it = 0; it < 2; ++it) {
    int r = rs + it * 64;
    cp16(&As[0][r * LDSW + sg], Ag + (size_t)r * KDIM + sg);
    cp16(&Bs[0][r * LDSW + sg], Bg + (size_t)r * KDIM + sg);
  }
  const int NK = KDIM / 32;
  for (int i = 0; i < NK; ++i) {
    async_wait0();          // own wave's tile-i async loads have landed
    __syncthreads();        // everyone's tile i visible; compute i-1 done
    if (i + 1 < NK) {       // stream tile i+1 into the other buffer
      int k0 = (i + 1) * 32, sel = (i + 1) & 1;
      for (int it = 0; it < 2; ++it) {
        int r = rs + it * 64;
        cp16(&As[sel][r * LDSW + sg], Ag + (size_t)r * KDIM + k0 + sg);
        cp16(&Bs[sel][r * LDSW + sg], Bg + (size_t)r * KDIM + k0 + sg);
      }
    }
    int sel = i & 1;
    v16h af[2], bf[4];
    for (int mi = 0; mi < 2; ++mi) af[mi] = ldsFrag(As[sel], wm * 32 + mi * 16, lane);
    for (int nj = 0; nj < 4; ++nj) bf[nj] = ldsFrag(Bs[sel], wn_ * 64 + nj * 16, lane);
    for (int mi = 0; mi < 2; ++mi)
      for (int nj = 0; nj < 4; ++nj)
        acc[mi][nj] = __builtin_amdgcn_wmma_f32_16x16x32_f16(
            false, af[mi], false, bf[nj], (short)0, acc[mi][nj], false, false);
  }
  // C/D layout: lane holds N=(lane&15); vgpr g -> M = g + (lane>=16 ? 8 : 0)
  int col = lane & 15, rb = (lane >> 4) << 3;
  float* Sb = S + ((size_t)b << 20);
  for (int mi = 0; mi < 2; ++mi)
    for (int nj = 0; nj < 4; ++nj) {
      int gm = pBase + wm * 32 + mi * 16 + rb;
      int gn = fBase + wn_ * 64 + nj * 16 + col;
      for (int g = 0; g < 8; ++g)
        Sb[(size_t)(gm + g) * 1024 + gn] = acc[mi][nj][g];
    }
}

// ------------------------------------------------ fuse passes (exact ref math)
__global__ void fuse1(const float* __restrict__ S, float* __restrict__ S1) {
  int idx = blockIdx.x * 256 + threadIdx.x;         // 8*1024*1024 total
  int b = idx >> 20, r = idx & 1048575;
  int p = r >> 10, f = r & 1023;
  const float* Sb = S + ((size_t)b << 20);
  float a = Sb[(size_t)p * 1024 + f];
  if (p > 0 && f > 0)       a += Sb[(size_t)(p - 1) * 1024 + (f - 1)];
  if (p < 1023 && f < 1023) a += Sb[(size_t)(p + 1) * 1024 + (f + 1)];
  S1[idx] = a;
}

#define TR(n) ((((n) & 31) << 5) | ((n) >> 5))
// S3T[b][f'][k'] = sum_d S1[b][tr(k'+d)][tr(f'+d)]  (the ref's transpose+fuse)
__global__ void fuse2t(const float* __restrict__ S1, float* __restrict__ S3T) {
  int idx = blockIdx.x * 256 + threadIdx.x;
  int b = idx >> 20, r = idx & 1048575;
  int f = r >> 10, k = r & 1023;
  const float* Sb = S1 + ((size_t)b << 20);
  float a = 0.f;
  for (int d = -1; d <= 1; ++d) {
    int kk = k + d, ff = f + d;
    if (kk >= 0 && kk < 1024 && ff >= 0 && ff < 1024)
      a += Sb[(size_t)TR(kk) * 1024 + TR(ff)];
  }
  S3T[idx] = a;
}

// softmax over k (the 1024 patch axis) with eq applied before AND after;
// zeroed (masked) entries still contribute exp() to the denominator, as in ref.
__global__ void __launch_bounds__(256) softmax_k(const float* __restrict__ S3T,
                                                 const float* __restrict__ eq,
                                                 half_t* __restrict__ PT) {
  __shared__ float red[256];
  int f = blockIdx.x, b = blockIdx.y, tid = threadIdx.x;
  size_t base = (size_t)(b * LPAT + f) * 1024;
  float t[4], lm = -3.4e38f;
  for (int q = 0; q < 4; ++q) {
    int k = q * 256 + tid;
    float tv = eq[k] * S3T[base + k];
    t[q] = tv; lm = fmaxf(lm, tv);
  }
  red[tid] = lm; __syncthreads();
  for (int s = 128; s > 0; s >>= 1) {
    if (tid < s) red[tid] = fmaxf(red[tid], red[tid + s]);
    __syncthreads();
  }
  float M = red[0]; __syncthreads();
  float e[4], ls = 0.f;
  for (int q = 0; q < 4; ++q) { e[q] = expf(10.f * (t[q] - M)); ls += e[q]; }
  red[tid] = ls; __syncthreads();
  for (int s = 128; s > 0; s >>= 1) {
    if (tid < s) red[tid] += red[tid + s];
    __syncthreads();
  }
  float inv = 1.f / red[0];
  for (int q = 0; q < 4; ++q) {
    int k = q * 256 + tid;
    PT[base + k] = (half_t)(eq[k] * e[q] * inv);
  }
}

// --------------------------- deconv GEMM: y += (1/4) RW(dy,dx)^T * P, pasted
// per (b,dy,dx): Out[c,f'] = sum_k bg[b,c,2(k/32)+dy-1,2(k%32)+dx-1] * P[k,f']
// output pixel (u,v) = (2i-1+dy, 2j-1+dx), f' = i*32+j; atomic add (overlap).
// B tile (P^T) streams via async-to-LDS; A tile is a bounds-checked gather.
__global__ void __launch_bounds__(256) deconv_gemm(const float* __restrict__ bg,
                                                   const half_t* __restrict__ PT,
                                                   float* __restrict__ y) {
  __shared__ __align__(16) half_t As[2][128 * LDSW];
  __shared__ __align__(16) half_t Bs[2][128 * LDSW];
  int tid = threadIdx.x, lane = tid & 31, wave = tid >> 5;
  int wm = wave & 3, wn_ = wave >> 2;
  int b = blockIdx.z;
  int dy = blockIdx.y >> 2, dx = blockIdx.y & 3;
  int fBase = blockIdx.x * 128;
  const half_t* Bg = PT + (size_t)(b * LPAT + fBase) * 1024;
  const float* bgb = bg + (size_t)b * 128 * 4096;
  int rs = tid >> 2, sg = (tid & 3) << 3;
  v8f acc[2][4];
  for (int mi = 0; mi < 2; ++mi)
    for (int nj = 0; nj < 4; ++nj) acc[mi][nj] = {};

#define DEC_STAGE_A(k0, sel)                                                   \
  {                                                                            \
    int row = 2 * ((k0) >> 5) + dy - 1;                                        \
    bool rok = (row >= 0) && (row < 64);                                       \
    for (int it = 0; it < 16; ++it) {                                          \
      int u = it * 256 + tid;                                                  \
      int c = u >> 5, kk = u & 31;                                             \
      int colg = 2 * kk + dx - 1;                                              \
      float v = 0.f;                                                           \
      if (rok && colg >= 0 && colg < 64)                                       \
        v = bgb[((size_t)c * 64 + row) * 64 + colg];                           \
      As[sel][c * LDSW + kk] = (half_t)v;                                      \
    }                                                                          \
  }
#define DEC_STAGE_B(k0, sel)                                                   \
  for (int it = 0; it < 2; ++it) {                                             \
    int r = rs + it * 64;                                                      \
    cp16(&Bs[sel][r * LDSW + sg], Bg + (size_t)r * 1024 + (k0) + sg);          \
  }

  DEC_STAGE_A(0, 0)
  DEC_STAGE_B(0, 0)
  const int NK = 1024 / 32;
  for (int i = 0; i < NK; ++i) {
    async_wait0();
    __syncthreads();
    if (i + 1 < NK) {
      int k0 = (i + 1) * 32, sel = (i + 1) & 1;
      DEC_STAGE_A(k0, sel)
      DEC_STAGE_B(k0, sel)
    }
    int sel = i & 1;
    v16h af[2], bf[4];
    for (int mi = 0; mi < 2; ++mi) af[mi] = ldsFrag(As[sel], wm * 32 + mi * 16, lane);
    for (int nj = 0; nj < 4; ++nj) bf[nj] = ldsFrag(Bs[sel], wn_ * 64 + nj * 16, lane);
    for (int mi = 0; mi < 2; ++mi)
      for (int nj = 0; nj < 4; ++nj)
        acc[mi][nj] = __builtin_amdgcn_wmma_f32_16x16x32_f16(
            false, af[mi], false, bf[nj], (short)0, acc[mi][nj], false, false);
  }
  int col = lane & 15, rb = (lane >> 4) << 3;
  for (int mi = 0; mi < 2; ++mi)
    for (int nj = 0; nj < 4; ++nj) {
      int n = fBase + wn_ * 64 + nj * 16 + col;
      int i = n >> 5, j = n & 31;
      int u = 2 * i - 1 + dy, v = 2 * j - 1 + dx;
      if (u >= 0 && u < 64 && v >= 0 && v < 64) {
        for (int g = 0; g < 8; ++g) {
          int c = wm * 32 + mi * 16 + rb + g;
          atomicAdd(&y[(((size_t)b * 128 + c) * 64 + u) * 64 + v],
                    0.25f * acc[mi][nj][g]);
        }
      }
    }
#undef DEC_STAGE_A
#undef DEC_STAGE_B
}

// ---------------------------------------------------------------- launcher
extern "C" void kernel_launch(void* const* d_in, const int* in_sizes, int n_in,
                              void* d_out, int out_size, void* d_ws, size_t ws_size,
                              hipStream_t stream) {
  (void)in_sizes; (void)n_in; (void)ws_size;
  const float* fg   = (const float*)d_in[0];   // (8,128,64,64)
  const float* bg   = (const float*)d_in[1];   // (8,128,64,64)
  const float* mask = (const float*)d_in[2];   // (1,1,64,64)
  float* y = (float*)d_out;                    // (8,128,64,64)

  char* ws = (char*)d_ws;                      // requires ~105 MB
  half_t* Wn  = (half_t*)(ws + 0);             // 8*1024*1152 f16 = 18.87 MB
  half_t* X   = (half_t*)(ws + 18874368);      // 18.87 MB
  float*  S   = (float*)(ws + 37748736);       // 8*1024*1024 f32 = 33.55 MB
  float*  S1  = (float*)(ws + 71303168);       // 33.55 MB
  float*  eqv = (float*)(ws + 104857600);      // 4 KB
  float*  S3T = (float*)(ws + 0);              // reuse Wn+X region (dead)
  half_t* PT  = (half_t*)(ws + 37748736);      // reuse S region (dead)

  zero_f32  <<<(out_size + 255) / 256, 256, 0, stream>>>(y, out_size);
  eq_kernel <<<4, 256, 0, stream>>>(mask, eqv);
  x_prep    <<<dim3(1024, 8), 128, 0, stream>>>(fg, X);
  wn_prep   <<<dim3(1024, 8), 128, 0, stream>>>(bg, Wn);
  score_gemm<<<dim3(8, 8, 8), 256, 0, stream>>>(Wn, X, S);
  fuse1     <<<32768, 256, 0, stream>>>(S, S1);
  fuse2t    <<<32768, 256, 0, stream>>>(S1, S3T);
  softmax_k <<<dim3(1024, 8), 256, 0, stream>>>(S3T, eqv, PT);
  deconv_gemm<<<dim3(8, 16, 8), 256, 0, stream>>>(bg, PT, y);
}
